// VO_16544214024389
// MI455X (gfx1250) — matrix operand order repaired and checked by reading, not compiled, not run on hardware
//
#include <hip/hip_runtime.h>
#include <hip/hip_bf16.h>
#include <math.h>

// ---------------------------------------------------------------------------
// MI455X (gfx1250) implementation of the iterative VO solver.
// Heavy GEMMs (~90 GFLOP total) run through v_wmma_f32_16x16x32_f16 (wave32,
// f32 accumulate). Weights are pre-transposed to N-major so both A and B
// WMMA fragments stage as contiguous b128 global->LDS traffic, software-
// pipelined (register-staged) across K tiles.
// ---------------------------------------------------------------------------

typedef __attribute__((ext_vector_type(16))) _Float16 v16h;
typedef __attribute__((ext_vector_type(8)))  float    v8f;

constexpr int cB  = 4;
constexpr int cN  = 2048;
constexpr int cT  = 4096;
constexpr int cD  = 256;
constexpr int cNN = cB * cN;        // 8192 nodes
constexpr int cE  = cB * cT * 6;    // 98304 edges
constexpr int cKP = 800;            // 772 padded to multiple of 32
constexpr int cITERS = 8;
constexpr float cNEG = 0.2f;

// bacc layout per batch: Hpp(36) | As(36) | bp(6) | rhs2(6)
constexpr int BACC = 84;

__device__ inline float sigf(float x) { return 1.f / (1.f + expf(-x)); }

// monotone float<->uint encoding for atomic float max
__device__ inline unsigned fenc(float f) {
  unsigned u = __float_as_uint(f);
  return (u & 0x80000000u) ? ~u : (u | 0x80000000u);
}
__device__ inline float fdec(unsigned k) {
  unsigned u = (k & 0x80000000u) ? (k & 0x7FFFFFFFu) : ~k;
  return __uint_as_float(u);
}

__device__ inline void so3exp(const float w[3], float R[9]) {
  float th2 = w[0]*w[0] + w[1]*w[1] + w[2]*w[2];
  float th  = sqrtf(fmaxf(th2, 1e-12f));
  float A   = (th2 > 1e-8f) ? (sinf(th) / th)          : (1.f  - th2 / 6.f);
  float Bc  = (th2 > 1e-8f) ? ((1.f - cosf(th)) / th2) : (0.5f - th2 / 24.f);
  float K[9]  = {0.f, -w[2], w[1],  w[2], 0.f, -w[0],  -w[1], w[0], 0.f};
  float K2[9];
  for (int i = 0; i < 3; ++i)
    for (int j = 0; j < 3; ++j) {
      float s = 0.f;
      for (int k = 0; k < 3; ++k) s += K[i*3+k] * K[k*3+j];
      K2[i*3+j] = s;
    }
  for (int i = 0; i < 3; ++i)
    for (int j = 0; j < 3; ++j)
      R[i*3+j] = (i == j ? 1.f : 0.f) + A * K[i*3+j] + Bc * K2[i*3+j];
}

__device__ inline void se3exp(const float xi[6], float R[9], float t[3]) {
  const float* v = xi;
  const float* w = xi + 3;
  float th2 = w[0]*w[0] + w[1]*w[1] + w[2]*w[2];
  float th  = sqrtf(fmaxf(th2, 1e-12f));
  float A   = (th2 > 1e-8f) ? (sinf(th) / th)          : (1.f    - th2 / 6.f);
  float Bc  = (th2 > 1e-8f) ? ((1.f - cosf(th)) / th2) : (0.5f   - th2 / 24.f);
  float Cc  = (th2 > 1e-8f) ? ((1.f - A) / th2)        : (1.f/6.f - th2 / 120.f);
  float K[9]  = {0.f, -w[2], w[1],  w[2], 0.f, -w[0],  -w[1], w[0], 0.f};
  float K2[9];
  for (int i = 0; i < 3; ++i)
    for (int j = 0; j < 3; ++j) {
      float s = 0.f;
      for (int k = 0; k < 3; ++k) s += K[i*3+k] * K[k*3+j];
      K2[i*3+j] = s;
    }
  float V[9];
  for (int i = 0; i < 3; ++i)
    for (int j = 0; j < 3; ++j) {
      R[i*3+j] = (i == j ? 1.f : 0.f) + A  * K[i*3+j] + Bc * K2[i*3+j];
      V[i*3+j] = (i == j ? 1.f : 0.f) + Bc * K[i*3+j] + Cc * K2[i*3+j];
    }
  for (int i = 0; i < 3; ++i)
    t[i] = V[i*3+0]*v[0] + V[i*3+1]*v[1] + V[i*3+2]*v[2];
}

// ---------------------------------------------------------------------------
// WMMA GEMM: C[M x Nc] (f32) = A[M x K] (f16, row-major, row stride K)
//                              * Wt[Nc x K] (f16, N-major, row stride K)
// Workgroup tile 128x64, 8 waves in a 4x2 grid, each wave = 2x2 wmma tiles.
// K tiles of 32 staged via registers (b128) -> LDS (b128), software-pipelined.
// ---------------------------------------------------------------------------
#define GBM 128
#define GBN 64
#define GBK 32

__global__ __launch_bounds__(256) void k_wmma_gemm(
    const _Float16* __restrict__ A, const _Float16* __restrict__ Wt,
    float* __restrict__ C, int M, int K, int Nc)
{
  __shared__ __align__(16) unsigned As[GBM * 16];   // [row][k/2]  (16 dwords)
  __shared__ __align__(16) unsigned Bs[GBN * 16];   // [n][k/2]    (16 dwords)

  const int tid  = threadIdx.x;
  const int lane = tid & 31;
  const int wid  = tid >> 5;
  const int wr   = wid >> 1;   // 0..3 (row group of 32)
  const int wc   = wid & 1;    // 0..1 (col group of 32)
  const int rowStart = blockIdx.x * GBM;
  const int colStart = blockIdx.y * GBN;

  // register staging: A tile = 512 uint4 (2/thread), B tile = 256 uint4 (1/thread)
  const int ar0 = (tid)        >> 2, aq0 = (tid)        & 3;
  const int ar1 = (tid + 256)  >> 2, aq1 = (tid + 256)  & 3;
  const int br  = tid >> 2,          bq  = tid & 3;
  const size_t a_str4 = (size_t)(K >> 3);  // uint4 per A row
  const size_t b_str4 = (size_t)(K >> 3);  // uint4 per Wt row

  const uint4* Ag4 = (const uint4*)(A  + (size_t)rowStart * K);
  const uint4* Bg4 = (const uint4*)(Wt + (size_t)colStart * K);

  uint4 pa0, pa1, pb;
  auto stage_load = [&](int k0) {
    const size_t kq = (size_t)(k0 >> 3);
    pa0 = Ag4[(size_t)ar0 * a_str4 + kq + aq0];
    pa1 = Ag4[(size_t)ar1 * a_str4 + kq + aq1];
    pb  = Bg4[(size_t)br  * b_str4 + kq + bq];
  };
  auto stage_store = [&]() {
    *(uint4*)&As[ar0 * 16 + aq0 * 4] = pa0;
    *(uint4*)&As[ar1 * 16 + aq1 * 4] = pa1;
    *(uint4*)&Bs[br  * 16 + bq  * 4] = pb;
  };

  v8f acc[2][2];
  for (int mi = 0; mi < 2; ++mi)
    for (int ni = 0; ni < 2; ++ni)
      for (int r = 0; r < 8; ++r) acc[mi][ni][r] = 0.f;

  const int mfrag  = lane & 15;
  const int a_koff = (lane >> 4) << 3;   // +8 dwords for hi half-wave (A contiguous)
  const int b_koff = (lane >> 4) << 3;   // +8 dwords for hi half-wave (B contiguous)

  stage_load(0);
  stage_store();

  for (int k0 = 0; k0 < K; k0 += GBK) {
    __syncthreads();
    const bool more = (k0 + GBK) < K;
    if (more) {
      stage_load(k0 + GBK);                       // overlap with WMMA below
      __builtin_prefetch(A + (size_t)(rowStart + (tid & 127)) * K + k0 + 2 * GBK, 0, 1);
    }

    union Frag { unsigned u[8]; v16h h; };
    Frag afrag[2], bfrag[2];
    // A fragment (lanes 0-15: K 0..15, lanes 16-31: K 16..31; contiguous dwords)
    #pragma unroll
    for (int mi = 0; mi < 2; ++mi) {
      int rb = wr * 32 + mi * 16 + mfrag;
      #pragma unroll
      for (int v = 0; v < 8; ++v) {
        int d = v + a_koff;
        afrag[mi].u[v] = As[rb * 16 + d];
      }
    }
    // B fragment (lanes 0-15: K 0..15, lanes 16-31: K 16..31; contiguous dwords)
    #pragma unroll
    for (int ni = 0; ni < 2; ++ni) {
      int nb = wc * 32 + ni * 16 + (lane & 15);
      #pragma unroll
      for (int v = 0; v < 8; ++v)
        bfrag[ni].u[v] = Bs[nb * 16 + v + b_koff];
    }
    #pragma unroll
    for (int mi = 0; mi < 2; ++mi)
      #pragma unroll
      for (int ni = 0; ni < 2; ++ni)
        acc[mi][ni] = __builtin_amdgcn_wmma_f32_16x16x32_f16(
            false, afrag[mi].h, false, bfrag[ni].h,
            (short)0, acc[mi][ni], false, false);
    __syncthreads();
    if (more) stage_store();
  }

  // store: C/D layout — VGPR r: M = r + (lane>=16 ? 8 : 0), N = lane&15
  const int moff = (lane >> 4) << 3;
  const int ncol = lane & 15;
  for (int mi = 0; mi < 2; ++mi)
    for (int ni = 0; ni < 2; ++ni) {
      int rb = rowStart + wr * 32 + mi * 16 + moff;
      int cb = colStart + wc * 32 + ni * 16 + ncol;
      #pragma unroll
      for (int r = 0; r < 8; ++r)
        C[(size_t)(rb + r) * Nc + cb] = acc[mi][ni][r];
    }
}

// ---------------------------------------------------------------------------
// Elementwise / graph kernels
// ---------------------------------------------------------------------------
__global__ void k_cvt16(const float* __restrict__ s, _Float16* __restrict__ d, int n) {
  int i = blockIdx.x * blockDim.x + threadIdx.x;
  if (i < n) d[i] = (_Float16)s[i];
}

// W (Krows x Ncols, row-major f32) -> Wt (Ncols x Kpad, row-major f16), zero pad
__global__ void k_wcvt(const float* __restrict__ s, _Float16* __restrict__ d,
                       int Krows, int Kpad, int Ncols) {
  int i = blockIdx.x * blockDim.x + threadIdx.x;
  if (i >= Kpad * Ncols) return;
  int n = i / Kpad, k = i % Kpad;
  d[i] = (k < Krows) ? (_Float16)s[(size_t)k * Ncols + n] : (_Float16)0.f;
}

__global__ void k_addgeo(const float* __restrict__ kpts, const float* __restrict__ pts3,
                         const float* __restrict__ Wgeo, float* __restrict__ x0) {
  int i = blockIdx.x * blockDim.x + threadIdx.x;
  if (i >= cNN * cD) return;
  int n = i / cD, d = i % cD;
  float g0 = kpts[n*2+0], g1 = kpts[n*2+1];
  float g2 = pts3[n*3+0], g3 = pts3[n*3+1], g4 = pts3[n*3+2];
  x0[i] += g0*Wgeo[0*cD+d] + g1*Wgeo[1*cD+d] + g2*Wgeo[2*cD+d]
         + g3*Wgeo[3*cD+d] + g4*Wgeo[4*cD+d];
}

__global__ void k_edges(const int* __restrict__ tri, const float* __restrict__ kpts,
                        int* __restrict__ esrc, int* __restrict__ edst,
                        float* __restrict__ ea) {
  int e = blockIdx.x * blockDim.x + threadIdx.x;
  if (e >= cE) return;
  int b = e / (cT * 6), rem = e % (cT * 6);
  int tr = rem / 6, wh = rem % 6;
  int vi = tri[(b*cT + tr)*3 + 0] + b*cN;
  int vj = tri[(b*cT + tr)*3 + 1] + b*cN;
  int vk = tri[(b*cT + tr)*3 + 2] + b*cN;
  int v3[3] = {vi, vj, vk};
  const int ssel[6] = {0,1,2,1,2,0};
  const int dsel[6] = {1,2,0,0,1,2};
  int s = v3[ssel[wh]], t = v3[dsel[wh]];
  esrc[e] = s; edst[e] = t;
  float dx = kpts[t*2+0] - kpts[s*2+0];
  float dy = kpts[t*2+1] - kpts[s*2+1];
  ea[e*3+0] = dx; ea[e*3+1] = dy; ea[e*3+2] = sqrtf(dx*dx + dy*dy);
}

__global__ void k_logit(const float* __restrict__ hs, const float* __restrict__ hd,
                        const float* __restrict__ We, const float* __restrict__ aatt,
                        const int* __restrict__ esrc, const int* __restrict__ edst,
                        const float* __restrict__ ea,
                        float* __restrict__ logit, unsigned* __restrict__ mx) {
  int e = blockIdx.x * blockDim.x + threadIdx.x;
  if (e >= cE) return;
  int s = esrc[e], t = edst[e];
  float a0 = ea[e*3+0], a1 = ea[e*3+1], a2 = ea[e*3+2];
  float acc = 0.f;
  for (int d = 0; d < cD; ++d) {
    float he = a0*We[d] + a1*We[cD+d] + a2*We[2*cD+d];
    float v = hs[s*cD+d] + hd[t*cD+d] + he;
    float lr = v > 0.f ? v : cNEG * v;
    acc += lr * aatt[d];
  }
  logit[e] = acc;
  atomicMax(&mx[t], fenc(acc));
}

__global__ void k_expden(const float* __restrict__ logit, const unsigned* __restrict__ mx,
                         const int* __restrict__ edst,
                         float* __restrict__ eexp, float* __restrict__ den,
                         float* __restrict__ ecnt) {
  int e = blockIdx.x * blockDim.x + threadIdx.x;
  if (e >= cE) return;
  int t = edst[e];
  float v = expf(logit[e] - fdec(mx[t]));
  eexp[e] = v;
  atomicAdd(&den[t], v);
  atomicAdd(&ecnt[t], 1.f);
}

__global__ void k_agg(const float* __restrict__ hs, const float* __restrict__ We,
                      const int* __restrict__ esrc, const int* __restrict__ edst,
                      const float* __restrict__ ea, const float* __restrict__ eexp,
                      const float* __restrict__ den, float* __restrict__ agg) {
  int e = blockIdx.x;
  int d = threadIdx.x;
  int s = esrc[e], t = edst[e];
  float alpha = eexp[e] / (den[t] + 1e-8f);
  float he = ea[e*3+0]*We[d] + ea[e*3+1]*We[cD+d] + ea[e*3+2]*We[2*cD+d];
  atomicAdd(&agg[t*cD+d], alpha * (hs[s*cD+d] + he));
}

__global__ void k_nf(const float* __restrict__ agg, const float* __restrict__ x0,
                     const float* __restrict__ mask, float* __restrict__ nf) {
  int i = blockIdx.x * blockDim.x + threadIdx.x;
  if (i >= cNN * cD) return;
  float v = agg[i] + x0[i];
  v = v > 0.f ? v : 0.f;
  nf[i] = v * mask[i / cD];
}

__global__ void k_pooled(const float* __restrict__ nf, float* __restrict__ pooled) {
  int i = blockIdx.x * blockDim.x + threadIdx.x;
  if (i >= cB * cD) return;
  int b = i / cD, d = i % cD;
  float s = 0.f;
  for (int n = 0; n < cN; ++n) s += nf[(size_t)(b*cN+n)*cD + d];
  pooled[i] = s / (float)cN;
}

__global__ void k_initpose(const float* __restrict__ pooled, const float* __restrict__ wrot,
                           float* __restrict__ Rm, float* __restrict__ tv) {
  int b = blockIdx.x * blockDim.x + threadIdx.x;
  if (b >= cB) return;
  float w[3];
  for (int c = 0; c < 3; ++c) {
    float s = 0.f;
    for (int d = 0; d < cD; ++d) s += pooled[b*cD+d] * wrot[d*3+c];
    w[c] = 0.1f * tanhf(s);
  }
  so3exp(w, Rm + b*9);
  tv[b*3+0] = 0.f; tv[b*3+1] = 0.f; tv[b*3+2] = 0.f;
}

__global__ void k_tri(const int* __restrict__ tri, const float* __restrict__ nf,
                      const float* __restrict__ wtri,
                      float* __restrict__ twsum, float* __restrict__ twcnt) {
  int q = blockIdx.x * blockDim.x + threadIdx.x;
  if (q >= cB * cT) return;
  int b = q / cT;
  int vi = tri[q*3+0] + b*cN, vj = tri[q*3+1] + b*cN, vk = tri[q*3+2] + b*cN;
  float s = 0.f;
  for (int d = 0; d < cD; ++d)
    s += (nf[(size_t)vi*cD+d] + nf[(size_t)vj*cD+d] + nf[(size_t)vk*cD+d]) * (1.f/3.f) * wtri[d];
  float tw = sigf(s);
  atomicAdd(&twsum[vi], tw); atomicAdd(&twcnt[vi], 1.f);
  atomicAdd(&twsum[vj], tw); atomicAdd(&twcnt[vj], 1.f);
  atomicAdd(&twsum[vk], tw); atomicAdd(&twcnt[vk], 1.f);
}

__global__ void k_twnode(const float* __restrict__ twsum, const float* __restrict__ twcnt,
                         float* __restrict__ twnode) {
  int i = blockIdx.x * blockDim.x + threadIdx.x;
  if (i < cNN) twnode[i] = twsum[i] / (twcnt[i] + 1e-8f);
}

__global__ void k_vpray(const float* __restrict__ nf, const float* __restrict__ wvp,
                        const float* __restrict__ kpts, const float* __restrict__ pts3,
                        const float* __restrict__ calib,
                        float* __restrict__ vp, float* __restrict__ ray,
                        float* __restrict__ depth) {
  int n = blockIdx.x * blockDim.x + threadIdx.x;
  if (n >= cNN) return;
  float s = 0.f;
  for (int d = 0; d < cD; ++d) s += nf[(size_t)n*cD+d] * wvp[d];
  vp[n] = sigf(s);
  int b = n / cN;
  float fx = calib[b*4+0], fy = calib[b*4+1], cx = calib[b*4+2], cy = calib[b*4+3];
  ray[n*3+0] = (kpts[n*2+0] - cx) / fx;
  ray[n*3+1] = (kpts[n*2+1] - cy) / fy;
  ray[n*3+2] = 1.f;
  depth[n] = pts3[n*3+2];
}

__global__ void k_proj(const float* __restrict__ depth, const float* __restrict__ ray,
                       const float* __restrict__ Rm, const float* __restrict__ tv,
                       const float* __restrict__ calib, const float* __restrict__ ktp1,
                       float* __restrict__ X, float* __restrict__ rres) {
  int n = blockIdx.x * blockDim.x + threadIdx.x;
  if (n >= cNN) return;
  int b = n / cN;
  float dep = depth[n];
  float x0 = dep * ray[n*3+0], x1 = dep * ray[n*3+1], x2 = dep * ray[n*3+2];
  const float* R = Rm + b*9;
  float p0 = R[0]*x0 + R[1]*x1 + R[2]*x2 + tv[b*3+0];
  float p1 = R[3]*x0 + R[4]*x1 + R[5]*x2 + tv[b*3+1];
  float p2 = R[6]*x0 + R[7]*x1 + R[8]*x2 + tv[b*3+2];
  float z = fmaxf(p2, 0.001f);
  float fx = calib[b*4+0], fy = calib[b*4+1], cx = calib[b*4+2], cy = calib[b*4+3];
  float ku = fx * p0 / z + cx;
  float kv = fy * p1 / z + cy;
  X[n*3+0] = x0; X[n*3+1] = x1; X[n*3+2] = x2;
  rres[n*2+0] = ktp1[n*2+0] - ku;
  rres[n*2+1] = ktp1[n*2+1] - kv;
}

__global__ void k_msgacc(const float* __restrict__ hW,
                         const int* __restrict__ esrc, const int* __restrict__ edst,
                         float* __restrict__ msg) {
  int e = blockIdx.x;
  int d = threadIdx.x;
  atomicAdd(&msg[(size_t)edst[e]*cD + d], hW[(size_t)esrc[e]*cD + d]);
}

__global__ void k_buildxh(const float* __restrict__ nf, const float* __restrict__ rres,
                          const float* __restrict__ twnode, const float* __restrict__ vp,
                          const float* __restrict__ msg, const float* __restrict__ ecnt,
                          const float* __restrict__ h, _Float16* __restrict__ xh) {
  int n = blockIdx.x;
  float inv = 1.f / (ecnt[n] + 1e-8f);
  for (int j = threadIdx.x; j < cKP; j += blockDim.x) {
    float v;
    if      (j < 256) v = nf[(size_t)n*cD + j];
    else if (j < 258) v = rres[n*2 + (j-256)];
    else if (j == 258) v = twnode[n];
    else if (j == 259) v = vp[n];
    else if (j < 516) v = msg[(size_t)n*cD + (j-260)] * inv;
    else if (j < 772) v = h[(size_t)n*cD + (j-516)];
    else v = 0.f;
    xh[(size_t)n*cKP + j] = (_Float16)v;
  }
}

__global__ void k_gate(const _Float16* __restrict__ xh, const float* __restrict__ rgl,
                       const float* __restrict__ h,
                       float* __restrict__ zg, _Float16* __restrict__ xn) {
  int n = blockIdx.x;
  int d = threadIdx.x;
  size_t i = (size_t)n*cD + d;
  zg[i] = sigf(zg[i]);                       // in place over zg_lin
  float rg = sigf(rgl[i]);
  xn[(size_t)n*cKP + 516 + d] = (_Float16)(rg * h[i]);
  for (int j = d; j < cKP; j += blockDim.x) {
    if (j < 516)       xn[(size_t)n*cKP + j] = xh[(size_t)n*cKP + j];
    else if (j >= 772) xn[(size_t)n*cKP + j] = (_Float16)0.f;
  }
}

__global__ void k_hup(const float* __restrict__ zg, const float* __restrict__ ngl,
                      float* __restrict__ h) {
  int i = blockIdx.x * blockDim.x + threadIdx.x;
  if (i >= cNN * cD) return;
  float z = zg[i];
  float ng = tanhf(ngl[i]);
  h[i] = (1.f - z) * h[i] + z * ng;
}

__global__ void k_post(const float* __restrict__ h, const float* __restrict__ Wconf,
                       const float* __restrict__ Wad, const float* __restrict__ mask,
                       float* __restrict__ conf, float* __restrict__ adep) {
  int n = blockIdx.x * blockDim.x + threadIdx.x;
  if (n >= cNN) return;
  float c0 = 0.f, c1 = 0.f, ad = 0.f;
  for (int d = 0; d < cD; ++d) {
    float hv = h[(size_t)n*cD + d];
    c0 += hv * Wconf[d*2+0];
    c1 += hv * Wconf[d*2+1];
    ad += hv * Wad[d];
  }
  float m = mask[n];
  conf[n*2+0] = sigf(c0) * m;
  conf[n*2+1] = sigf(c1) * m;
  adep[n] = sigf(ad);
}

__global__ void k_hmean(const float* __restrict__ h, float* __restrict__ hmean) {
  int i = blockIdx.x * blockDim.x + threadIdx.x;
  if (i >= cB * cD) return;
  int b = i / cD, d = i % cD;
  float s = 0.f;
  for (int n = 0; n < cN; ++n) s += h[(size_t)(b*cN+n)*cD + d];
  hmean[i] = s / (float)cN;
}

__global__ void k_ap(const float* __restrict__ hmean, const float* __restrict__ Wap,
                     float* __restrict__ apv) {
  int b = blockIdx.x * blockDim.x + threadIdx.x;
  if (b >= cB) return;
  float s = 0.f;
  for (int d = 0; d < cD; ++d) s += hmean[b*cD+d] * Wap[d];
  apv[b] = sigf(s);
}

__global__ void k_accum(const float* __restrict__ X, const float* __restrict__ ray,
                        const float* __restrict__ rres, const float* __restrict__ conf,
                        const float* __restrict__ calib, const float* __restrict__ loglm,
                        float* __restrict__ Hpd, float* __restrict__ Hdd,
                        float* __restrict__ bd, float* __restrict__ bacc) {
  int n = blockIdx.x * blockDim.x + threadIdx.x;
  if (n >= cNN) return;
  int b = n / cN;
  float lm = expf(loglm[0]);
  float fx = calib[b*4+0], fy = calib[b*4+1];
  float x0 = X[n*3+0], x1 = X[n*3+1], x2 = X[n*3+2];
  float zs = fmaxf(x2, 0.001f), iz = 1.f / zs;
  float du[3] = {fx*iz, 0.f, -fx*x0*iz*iz};
  float dv[3] = {0.f, fy*iz, -fy*x1*iz*iz};
  // M = -skew(X)
  float M[9] = {0.f, x2, -x1,  -x2, 0.f, x0,  x1, -x0, 0.f};
  float Jp[2][6], Jd[2];
  const float* rw = ray + n*3;
  for (int a = 0; a < 2; ++a) {
    const float* duv = a ? dv : du;
    Jp[a][0] = duv[0]; Jp[a][1] = duv[1]; Jp[a][2] = duv[2];
    for (int j = 0; j < 3; ++j)
      Jp[a][3+j] = duv[0]*M[0*3+j] + duv[1]*M[1*3+j] + duv[2]*M[2*3+j];
    Jd[a] = duv[0]*rw[0] + duv[1]*rw[1] + duv[2]*rw[2];
  }
  float w0 = conf[n*2+0], w1 = conf[n*2+1];
  float r0 = rres[n*2+0], r1 = rres[n*2+1];
  float hpd[6];
  for (int i = 0; i < 6; ++i)
    hpd[i] = w0*Jp[0][i]*Jd[0] + w1*Jp[1][i]*Jd[1];
  float hdd = w0*Jd[0]*Jd[0] + w1*Jd[1]*Jd[1] + lm;
  float bdn = w0*Jd[0]*r0 + w1*Jd[1]*r1;
  for (int i = 0; i < 6; ++i) Hpd[n*6+i] = hpd[i];
  Hdd[n] = hdd;
  bd[n] = bdn;
  float ih = 1.f / hdd;
  float* acc = bacc + b * BACC;
  for (int i = 0; i < 6; ++i) {
    for (int j = 0; j < 6; ++j) {
      atomicAdd(&acc[i*6+j],      w0*Jp[0][i]*Jp[0][j] + w1*Jp[1][i]*Jp[1][j]);
      atomicAdd(&acc[36 + i*6+j], hpd[i]*hpd[j]*ih);
    }
    atomicAdd(&acc[72 + i], w0*Jp[0][i]*r0 + w1*Jp[1][i]*r1);
    atomicAdd(&acc[78 + i], hpd[i]*bdn*ih);
  }
}

__global__ void k_solve(const float* __restrict__ bacc, const float* __restrict__ loglm,
                        const float* __restrict__ apv,
                        float* __restrict__ dpv, float* __restrict__ Rm,
                        float* __restrict__ tv) {
  int b = threadIdx.x;
  if (b >= cB) return;
  const float* acc = bacc + b * BACC;
  float lm = expf(loglm[0]);
  float S[36], rhs[6], dp[6];
  for (int i = 0; i < 6; ++i) {
    for (int j = 0; j < 6; ++j)
      S[i*6+j] = acc[i*6+j] + (i == j ? lm : 0.f) - acc[36 + i*6+j];
    rhs[i] = acc[72+i] - acc[78+i];
  }
  // Gaussian elimination with partial pivoting
  for (int c = 0; c < 6; ++c) {
    int p = c; float mv = fabsf(S[c*6+c]);
    for (int r = c+1; r < 6; ++r) { float a = fabsf(S[r*6+c]); if (a > mv) { mv = a; p = r; } }
    if (p != c) {
      for (int j = 0; j < 6; ++j) { float t = S[c*6+j]; S[c*6+j] = S[p*6+j]; S[p*6+j] = t; }
      float t = rhs[c]; rhs[c] = rhs[p]; rhs[p] = t;
    }
    float inv = 1.f / S[c*6+c];
    for (int r = c+1; r < 6; ++r) {
      float f = S[r*6+c] * inv;
      for (int j = c; j < 6; ++j) S[r*6+j] -= f * S[c*6+j];
      rhs[r] -= f * rhs[c];
    }
  }
  for (int c = 5; c >= 0; --c) {
    float s = rhs[c];
    for (int j = c+1; j < 6; ++j) s -= S[c*6+j] * dp[j];
    dp[c] = s / S[c*6+c];
  }
  for (int i = 0; i < 6; ++i) dpv[b*6+i] = dp[i];
  float xi[6];
  float ap = apv[b];
  for (int i = 0; i < 6; ++i) xi[i] = ap * dp[i];
  float dR[9], dt[3];
  se3exp(xi, dR, dt);
  float told[3] = {tv[b*3+0], tv[b*3+1], tv[b*3+2]};
  float Rold[9];
  for (int i = 0; i < 9; ++i) Rold[i] = Rm[b*9+i];
  for (int i = 0; i < 3; ++i)
    tv[b*3+i] = dR[i*3+0]*told[0] + dR[i*3+1]*told[1] + dR[i*3+2]*told[2] + dt[i];
  for (int i = 0; i < 3; ++i)
    for (int j = 0; j < 3; ++j)
      Rm[b*9 + i*3+j] = dR[i*3+0]*Rold[0*3+j] + dR[i*3+1]*Rold[1*3+j] + dR[i*3+2]*Rold[2*3+j];
}

__global__ void k_depth(const float* __restrict__ dpv, const float* __restrict__ Hpd,
                        const float* __restrict__ Hdd, const float* __restrict__ bd,
                        const float* __restrict__ adep, float* __restrict__ depth) {
  int n = blockIdx.x * blockDim.x + threadIdx.x;
  if (n >= cNN) return;
  int b = n / cN;
  float s = bd[n];
  for (int i = 0; i < 6; ++i) s -= Hpd[n*6+i] * dpv[b*6+i];
  float dd = s / Hdd[n];
  depth[n] = fmaxf(depth[n] + adep[n] * dd, 0.05f);
}

__global__ void k_out(const float* __restrict__ Rm, const float* __restrict__ tv,
                      const float* __restrict__ depth, float* __restrict__ out) {
  int i = blockIdx.x * blockDim.x + threadIdx.x;
  const int stride = 12 + cN;
  if (i >= cB * stride) return;
  int b = i / stride, j = i % stride;
  float v;
  if (j < 9)       v = Rm[b*9 + j];
  else if (j < 12) v = tv[b*3 + (j-9)];
  else             v = depth[b*cN + (j-12)];
  out[i] = v;
}

// ---------------------------------------------------------------------------
// Host launch
// ---------------------------------------------------------------------------
extern "C" void kernel_launch(void* const* d_in, const int* in_sizes, int n_in,
                              void* d_out, int out_size, void* d_ws, size_t ws_size,
                              hipStream_t stream)
{
  (void)in_sizes; (void)n_in; (void)out_size; (void)ws_size;
  // setup_inputs() insertion order; params dict flattened in insertion order.
  const float* kpts   = (const float*)d_in[0];
  const float* pts3   = (const float*)d_in[1];
  const float* descs  = (const float*)d_in[2];
  const float* ktp1   = (const float*)d_in[3];
  const float* mask   = (const float*)d_in[4];
  const float* calib  = (const float*)d_in[5];
  const float* W_desc = (const float*)d_in[6];
  const float* W_geo  = (const float*)d_in[7];
  const float* W_s    = (const float*)d_in[8];
  const float* W_d    = (const float*)d_in[9];
  const float* W_e    = (const float*)d_in[10];
  const float* a_att  = (const float*)d_in[11];
  const float* w_rot  = (const float*)d_in[12];
  const float* w_tri  = (const float*)d_in[13];
  const float* w_vp   = (const float*)d_in[14];
  const float* W_msg  = (const float*)d_in[15];
  const float* Wz     = (const float*)d_in[16];
  const float* Wr     = (const float*)d_in[17];
  const float* Wn     = (const float*)d_in[18];
  const float* W_conf = (const float*)d_in[19];
  const float* W_ap   = (const float*)d_in[20];
  const float* W_ad   = (const float*)d_in[21];
  const float* loglm  = (const float*)d_in[22];
  const int*   tri    = (const int*)d_in[23];
  float* out = (float*)d_out;

  char* ws = (char*)d_ws;
  size_t off = 0;
  auto alloc = [&](size_t bytes) -> void* {
    void* p = ws + off;
    off += (bytes + 255) & ~(size_t)255;
    return p;
  };

  _Float16* descs16 = (_Float16*)alloc((size_t)cNN*cD*2);
  float*    x0      = (float*)   alloc((size_t)cNN*cD*4);
  _Float16* x016    = (_Float16*)alloc((size_t)cNN*cD*2);
  float*    hs      = (float*)   alloc((size_t)cNN*cD*4);
  float*    hd      = (float*)   alloc((size_t)cNN*cD*4);
  _Float16* Wd16    = (_Float16*)alloc((size_t)cD*cD*2);     // transposed NxK
  _Float16* Ws16    = (_Float16*)alloc((size_t)cD*cD*2);
  _Float16* Wdd16   = (_Float16*)alloc((size_t)cD*cD*2);
  _Float16* Wm16    = (_Float16*)alloc((size_t)cD*cD*2);
  _Float16* Wz16    = (_Float16*)alloc((size_t)cD*cKP*2);    // transposed NxKP
  _Float16* Wr16    = (_Float16*)alloc((size_t)cD*cKP*2);
  _Float16* Wn16    = (_Float16*)alloc((size_t)cD*cKP*2);
  int*      esrc    = (int*)     alloc((size_t)cE*4);
  int*      edst    = (int*)     alloc((size_t)cE*4);
  float*    ea      = (float*)   alloc((size_t)cE*3*4);
  float*    logit   = (float*)   alloc((size_t)cE*4);
  float*    eexp    = (float*)   alloc((size_t)cE*4);
  unsigned* mx      = (unsigned*)alloc((size_t)cNN*4);
  float*    den     = (float*)   alloc((size_t)cNN*4);
  float*    ecnt    = (float*)   alloc((size_t)cNN*4);
  float*    agg     = (float*)   alloc((size_t)cNN*cD*4);
  float*    nf      = (float*)   alloc((size_t)cNN*cD*4);
  float*    pooled  = (float*)   alloc((size_t)cB*cD*4);
  float*    Rm      = (float*)   alloc((size_t)cB*9*4);
  float*    tv      = (float*)   alloc((size_t)cB*3*4);
  float*    twsum   = (float*)   alloc((size_t)cNN*4);
  float*    twcnt   = (float*)   alloc((size_t)cNN*4);
  float*    twnode  = (float*)   alloc((size_t)cNN*4);
  float*    vp      = (float*)   alloc((size_t)cNN*4);
  float*    ray     = (float*)   alloc((size_t)cNN*3*4);
  float*    depth   = (float*)   alloc((size_t)cNN*4);
  float*    h       = (float*)   alloc((size_t)cNN*cD*4);
  _Float16* h16     = (_Float16*)alloc((size_t)cNN*cD*2);
  float*    hW      = (float*)   alloc((size_t)cNN*cD*4);
  float*    msg     = (float*)   alloc((size_t)cNN*cD*4);
  _Float16* xh16    = (_Float16*)alloc((size_t)cNN*cKP*2);
  _Float16* xn16    = (_Float16*)alloc((size_t)cNN*cKP*2);
  float*    zg      = (float*)   alloc((size_t)cNN*cD*4);
  float*    rgl     = (float*)   alloc((size_t)cNN*cD*4);
  float*    ngl     = (float*)   alloc((size_t)cNN*cD*4);
  float*    Xb      = (float*)   alloc((size_t)cNN*3*4);
  float*    rres    = (float*)   alloc((size_t)cNN*2*4);
  float*    conf    = (float*)   alloc((size_t)cNN*2*4);
  float*    adep    = (float*)   alloc((size_t)cNN*4);
  float*    Hpd     = (float*)   alloc((size_t)cNN*6*4);
  float*    Hdd     = (float*)   alloc((size_t)cNN*4);
  float*    bd      = (float*)   alloc((size_t)cNN*4);
  float*    bacc    = (float*)   alloc((size_t)cB*BACC*4);
  float*    hmean   = (float*)   alloc((size_t)cB*cD*4);
  float*    apv     = (float*)   alloc((size_t)cB*4);
  float*    dpv     = (float*)   alloc((size_t)cB*6*4);

  auto gemm = [&](const _Float16* A, const _Float16* Wt, float* C, int M, int K, int Nc) {
    k_wmma_gemm<<<dim3(M/GBM, Nc/GBN), dim3(256), 0, stream>>>(A, Wt, C, M, K, Nc);
  };
  auto g1 = [](int n) { return dim3((n + 255) / 256); };

  // ---- weight / input conversions (weights transposed to N-major) ----
  k_cvt16<<<g1(cNN*cD), 256, 0, stream>>>(descs, descs16, cNN*cD);
  k_wcvt<<<g1(cD*cD),  256, 0, stream>>>(W_desc, Wd16,  cD, cD, cD);
  k_wcvt<<<g1(cD*cD),  256, 0, stream>>>(W_s,    Ws16,  cD, cD, cD);
  k_wcvt<<<g1(cD*cD),  256, 0, stream>>>(W_d,    Wdd16, cD, cD, cD);
  k_wcvt<<<g1(cD*cD),  256, 0, stream>>>(W_msg,  Wm16,  cD, cD, cD);
  k_wcvt<<<g1(cD*cKP), 256, 0, stream>>>(Wz, Wz16, 772, cKP, cD);
  k_wcvt<<<g1(cD*cKP), 256, 0, stream>>>(Wr, Wr16, 772, cKP, cD);
  k_wcvt<<<g1(cD*cKP), 256, 0, stream>>>(Wn, Wn16, 772, cKP, cD);

  hipMemsetAsync(mx,    0, (size_t)cNN*4, stream);
  hipMemsetAsync(den,   0, (size_t)cNN*4, stream);
  hipMemsetAsync(ecnt,  0, (size_t)cNN*4, stream);
  hipMemsetAsync(agg,   0, (size_t)cNN*cD*4, stream);
  hipMemsetAsync(twsum, 0, (size_t)cNN*4, stream);
  hipMemsetAsync(twcnt, 0, (size_t)cNN*4, stream);
  hipMemsetAsync(h,     0, (size_t)cNN*cD*4, stream);

  // ---- graph attention front-end ----
  k_edges<<<g1(cE), 256, 0, stream>>>(tri, kpts, esrc, edst, ea);
  gemm(descs16, Wd16, x0, cNN, cD, cD);
  k_addgeo<<<g1(cNN*cD), 256, 0, stream>>>(kpts, pts3, W_geo, x0);
  k_cvt16<<<g1(cNN*cD), 256, 0, stream>>>(x0, x016, cNN*cD);
  gemm(x016, Ws16,  hs, cNN, cD, cD);
  gemm(x016, Wdd16, hd, cNN, cD, cD);
  k_logit<<<g1(cE), 256, 0, stream>>>(hs, hd, W_e, a_att, esrc, edst, ea, logit, mx);
  k_expden<<<g1(cE), 256, 0, stream>>>(logit, mx, edst, eexp, den, ecnt);
  k_agg<<<dim3(cE), 256, 0, stream>>>(hs, W_e, esrc, edst, ea, eexp, den, agg);
  k_nf<<<g1(cNN*cD), 256, 0, stream>>>(agg, x0, mask, nf);
  k_pooled<<<g1(cB*cD), 256, 0, stream>>>(nf, pooled);
  k_initpose<<<1, 64, 0, stream>>>(pooled, w_rot, Rm, tv);
  k_tri<<<g1(cB*cT), 256, 0, stream>>>(tri, nf, w_tri, twsum, twcnt);
  k_twnode<<<g1(cNN), 256, 0, stream>>>(twsum, twcnt, twnode);
  k_vpray<<<g1(cNN), 256, 0, stream>>>(nf, w_vp, kpts, pts3, calib, vp, ray, depth);

  // ---- iterative solver ----
  for (int it = 0; it < cITERS; ++it) {
    k_proj<<<g1(cNN), 256, 0, stream>>>(depth, ray, Rm, tv, calib, ktp1, Xb, rres);
    hipMemsetAsync(msg,  0, (size_t)cNN*cD*4, stream);
    hipMemsetAsync(bacc, 0, (size_t)cB*BACC*4, stream);
    k_cvt16<<<g1(cNN*cD), 256, 0, stream>>>(h, h16, cNN*cD);
    gemm(h16, Wm16, hW, cNN, cD, cD);
    k_msgacc<<<dim3(cE), 256, 0, stream>>>(hW, esrc, edst, msg);
    k_buildxh<<<dim3(cNN), 256, 0, stream>>>(nf, rres, twnode, vp, msg, ecnt, h, xh16);
    gemm(xh16, Wz16, zg,  cNN, cKP, cD);
    gemm(xh16, Wr16, rgl, cNN, cKP, cD);
    k_gate<<<dim3(cNN), 256, 0, stream>>>(xh16, rgl, h, zg, xn16);
    gemm(xn16, Wn16, ngl, cNN, cKP, cD);
    k_hup<<<g1(cNN*cD), 256, 0, stream>>>(zg, ngl, h);
    k_post<<<g1(cNN), 256, 0, stream>>>(h, W_conf, W_ad, mask, conf, adep);
    k_hmean<<<g1(cB*cD), 256, 0, stream>>>(h, hmean);
    k_ap<<<1, 64, 0, stream>>>(hmean, W_ap, apv);
    k_accum<<<g1(cNN), 256, 0, stream>>>(Xb, ray, rres, conf, calib, loglm,
                                         Hpd, Hdd, bd, bacc);
    k_solve<<<1, 64, 0, stream>>>(bacc, loglm, apv, dpv, Rm, tv);
    k_depth<<<g1(cNN), 256, 0, stream>>>(dpv, Hpd, Hdd, bd, adep, depth);
  }

  k_out<<<g1(cB*(12+cN)), 256, 0, stream>>>(Rm, tv, depth, out);
}
